// FeatureExtractor_82454782148838
// MI455X (gfx1250) — matrix operand order repaired and checked by reading, not compiled, not run on hardware
//
#include <hip/hip_runtime.h>
#include <hip/hip_bf16.h>
#include <math.h>
#include <stdint.h>

// ---------------------------------------------------------------------------
// CDNA5 / gfx1250 wave32 WMMA fragments
typedef __attribute__((ext_vector_type(16))) _Float16 v16h;
typedef __attribute__((ext_vector_type(8)))  float    v8f;
typedef __attribute__((ext_vector_type(4)))  unsigned int u32x4;
typedef __attribute__((ext_vector_type(8)))  int      i32x8;
typedef __attribute__((ext_vector_type(4)))  int      i32x4;

#if defined(__has_builtin)
#  if __has_builtin(__builtin_amdgcn_tensor_load_to_lds)
#    define USE_TDM 1
#  endif
#endif
#ifndef USE_TDM
#  define USE_TDM 0
#endif

// A-matrix (16-bit, 16x32): a lane's v16h = two groups of 8 consecutive K,
// group g covering K = 32*ks + 16*g + 8*(lane>=16) + {0..7}  (ISA 7.12.2).
// B-matrix (16-bit, 32x16): lane = N(mod 16); same K order as A: e<8 -> K=8*(lane>=16)+e, e>=8 -> K=16+8*(lane>=16)+(e-8).
__device__ __forceinline__ int b_kidx(int lane, int e) {
  return ((e & 8) ? 16 : 0) + ((lane & 16) ? 8 : 0) + (e & 7);
}

#define WMMA_F16(A, B, C) \
  __builtin_amdgcn_wmma_f32_16x16x32_f16(false, (A), false, (B), (short)0, (C), false, false)

// ---------------------------------------------------------------------------
// Weight pack kernels: fragment-ordered f16, zero-filled at padded-K holes, so
// the GEMM kernels load each B fragment as one contiguous 32B chunk per lane.
// conv1: padded K' = 12*64 = 768  -> wpack1[24 ksteps][4 cg][32 lane][16 e]
// conv2: padded K' = 64*16 = 1024 -> wpack2[32 ksteps][8 cg][32 lane][16 e]
// ---------------------------------------------------------------------------
__global__ __launch_bounds__(256)
void pack_w1_kernel(const float* __restrict__ w, _Float16* __restrict__ wpack1) {
  int idx = blockIdx.x * 256 + threadIdx.x;          // 49152 total
  int e = idx & 15, lane = (idx >> 4) & 31, cg = (idx >> 9) & 3, ks = idx >> 11;
  int kb = ks * 32 + b_kidx(lane, e);
  int ic = kb >> 6, ky = (kb >> 3) & 7, kx = kb & 7;
  int oc = cg * 16 + (lane & 15);
  float v = (ky < 7 && kx < 7) ? w[oc * 588 + ic * 49 + ky * 7 + kx] : 0.f;
  wpack1[idx] = (_Float16)v;
}

__global__ __launch_bounds__(256)
void pack_w2_kernel(const float* __restrict__ w, _Float16* __restrict__ wpack2) {
  int idx = blockIdx.x * 256 + threadIdx.x;          // 131072 total
  int e = idx & 15, lane = (idx >> 4) & 31, cg = (idx >> 9) & 7, ks = idx >> 12;
  int kb = ks * 32 + b_kidx(lane, e);
  int ic = kb >> 4, ky = (kb >> 2) & 3, kx = kb & 3;
  int oc = cg * 16 + (lane & 15);
  float v = (ky < 3 && kx < 3) ? w[oc * 576 + ic * 9 + ky * 3 + kx] : 0.f;
  wpack2[idx] = (_Float16)v;
}

// ---------------------------------------------------------------------------
// Kernel 1: conv1 7x7 (12->64, pad 3) + ReLU + 2x2/2 maxpool, implicit GEMM.
// Workgroup = (b, 4-row strip, 32-col group); input patch 12x10x40 staged in
// LDS as f16 with zeroed halo + zero rows. Fragment build is branchless:
// each 8-element group is one (ic,ky) row, kx=0..6 contiguous, kx=7 == 0,
// row invalid (ky==7) -> one select to the zero slot.
// Each wave: M = its 4x4 pixel block, reuses A across all 4 channel groups.
// Output pooled[8][64][64][64] f16 (feeds conv2's TDM load).
// ---------------------------------------------------------------------------
__global__ __launch_bounds__(256)
void conv1_pool_kernel(const float* __restrict__ images,
                       const _Float16* __restrict__ wpack1,
                       const float* __restrict__ bias,
                       _Float16* __restrict__ pooled) {
  __shared__ _Float16 patch[12 * 10 * 40 + 16];      // zero slot @ 4800
  const int tid = threadIdx.x;
  const int cx = blockIdx.x & 3;
  const int sy = (blockIdx.x >> 2) & 31;
  const int b  = blockIdx.x >> 7;
  const int py0 = sy * 4, px0 = cx * 32;

  // stage: rows py0-3..py0+6, cols px0-3..px0+34 (cols 38,39 stay zero)
  for (int idx = tid; idx < 4800; idx += 256) {
    int ic  = idx / 400;
    int r   = idx - ic * 400;
    int row = r / 40, col = r - (r / 40) * 40;
    int iy = py0 + row - 3, ix = px0 + col - 3;
    float v = 0.f;
    if ((unsigned)iy < 128u && (unsigned)ix < 128u && col < 38)
      v = images[((b * 12 + ic) * 128 + iy) * 128 + ix];
    patch[idx] = (_Float16)v;
  }
  if (tid < 16) patch[4800 + tid] = (_Float16)0.f;
  __syncthreads();

  const int lane = tid & 31;
  const int wv   = tid >> 5;                         // pixel block 0..7
  const int h    = lane >> 4;                        // half-wave
  const int m    = lane & 15;
  const int dy   = m >> 2, dx = m & 3;
  const int xl   = wv * 4 + dx;                      // 0..31 within strip
  const v16h* wp = (const v16h*)wpack1;

  v8f acc0 = {}, acc1 = {}, acc2 = {}, acc3 = {};
  for (int ks = 0; ks < 24; ++ks) {                  // K' = 768
    v16h a;
#pragma unroll
    for (int g = 0; g < 2; ++g) {
      int kb = ks * 32 + 16 * g + 8 * h;             // group's K base
      int ic = kb >> 6, ky = (kb >> 3) & 7;
      int ra = (ky < 7) ? (ic * 400 + (dy + ky) * 40 + xl) : 4800;
#pragma unroll
      for (int kx = 0; kx < 7; ++kx)
        a[8 * g + kx] = patch[ra + kx];
      a[8 * g + 7] = (_Float16)0.f;                  // kx=7 is always padding
    }
    const int base = (ks * 4) * 32 + lane;
    v16h b0 = wp[base +  0], b1 = wp[base + 32];
    v16h b2 = wp[base + 64], b3 = wp[base + 96];
    acc0 = WMMA_F16(a, b0, acc0);
    acc1 = WMMA_F16(a, b1, acc1);
    acc2 = WMMA_F16(a, b2, acc2);
    acc3 = WMMA_F16(a, b3, acc3);
  }

  // C layout: lane<16 -> N=lane, M=r ; lane>=16 -> N=lane-16, M=8+r.
  const int n  = lane & 15;
  const int Y  = (py0 >> 1) + h;                     // pool row in this half-wave
  const int Xb = (px0 + wv * 4) >> 1;
#pragma unroll
  for (int j = 0; j < 4; ++j) {
    v8f acc = (j == 0) ? acc0 : (j == 1) ? acc1 : (j == 2) ? acc2 : acc3;
    int ch = j * 16 + n;
    float bz = bias[ch];
#pragma unroll
    for (int qx = 0; qx < 2; ++qx) {                 // pool needs r = 2qx+{0,1,4,5}
      float v0 = fmaxf(acc[2 * qx + 0] + bz, 0.f);
      float v1 = fmaxf(acc[2 * qx + 1] + bz, 0.f);
      float v2 = fmaxf(acc[2 * qx + 4] + bz, 0.f);
      float v3 = fmaxf(acc[2 * qx + 5] + bz, 0.f);
      float p  = fmaxf(fmaxf(v0, v1), fmaxf(v2, v3));
      pooled[((b * 64 + ch) * 64 + Y) * 64 + (Xb + qx)] = (_Float16)p;
    }
  }
}

// ---------------------------------------------------------------------------
// Kernel 2: conv2 3x3 (64->128, pad 1) + ReLU + position embedding.
// Workgroup = (b, y). LDS tile = pooled[b][ic][y-1..y+1][x] f16 laid out as
// [192 rows][66] with a 2-half x-halo: the TDM pad feature (pad after every
// 32 DWORDs, 1 DWORD pad) produces exactly this layout; pad columns and the
// 2 leading halo halves are pre-zeroed, so the fragment build needs NO ix
// bounds check. Per group: sub-row ky=2h always valid; sub-row ky=2h+1 valid
// iff h==0 (one half-wave-uniform select); kx=3 is always padding (=0).
// Output channel-last x2[b][y][x][c] f32.
// ---------------------------------------------------------------------------
#define T2_DATA   2                                   // leading halo halves
#define T2_ZSLOT  (2 + 192 * 66)                      // 12674
__global__ __launch_bounds__(256)
void conv2_emb_kernel(const _Float16* __restrict__ pooled,
                      const _Float16* __restrict__ wpack2,
                      const float* __restrict__ bias,
                      const float* __restrict__ pe_w,
                      const float* __restrict__ pe_b,
                      float* __restrict__ x2) {
  __shared__ _Float16 tile2[T2_ZSLOT + 14];           // 12688 halves
  const int tid = threadIdx.x;
  const int y = blockIdx.x & 63;
  const int b = blockIdx.x >> 6;
  const _Float16* src = pooled + (size_t)b * 64 * 4096;

  // pre-zero halo: 2 leading halves, 2 pad halves per row, zero slot.
  if (tid < 2)  tile2[tid] = (_Float16)0.f;
  if (tid < 14) tile2[T2_ZSLOT + tid] = (_Float16)0.f;
  if (tid < 192) {
    tile2[T2_DATA + tid * 66 + 64] = (_Float16)0.f;
    tile2[T2_DATA + tid * 66 + 65] = (_Float16)0.f;
  }

#if USE_TDM
  if (tid < 32) {                                     // one wave drives the TDM
    uint32_t lds = (uint32_t)(uintptr_t)&tile2[T2_DATA];
    uint64_t ga  = (uint64_t)((int64_t)(uintptr_t)src + (int64_t)(y - 1) * 128);
    // D# group0: count=1 | lds_addr | global_addr | type=2
    u32x4 g0 = { 1u, lds, (uint32_t)ga,
                 (uint32_t)((ga >> 32) & 0x1FFFFFFu) | (2u << 30) };
    // D# group1: data_size=2B; pad 1 DWORD after every 32 DWORDs;
    // tensor=[4096][64]; tile=[192][64]; tensor_dim0_stride=4096
    i32x8 g1 = { (int)((1u << 16) | (1u << 20) | (4u << 22)),
                 (int)(4096u << 16),     // tensor_dim0[15:0]
                 (int)(64u << 16),       // tensor_dim0 hi=0 | tensor_dim1 lo
                 (int)(192u << 16),      // tensor_dim1 hi=0 | tile_dim0=192
                 64,                     // tile_dim1=64 | tile_dim2=0
                 4096,                   // tensor_dim0_stride[31:0]
                 0, 0 };                 // stride0 hi, stride1 (unused, 2D)
    i32x4 gz4 = { 0, 0, 0, 0 };          // groups 2/3 unused for 2D tile
    i32x8 gz8 = { 0, 0, 0, 0, 0, 0, 0, 0 };
    __builtin_amdgcn_tensor_load_to_lds(g0, g1, gz4, gz4, gz8, 0);
    __builtin_amdgcn_s_wait_tensorcnt(0);
  }
#else
  for (int idx = tid; idx < 64 * 192; idx += 256) {
    int r  = idx >> 6;                                // ic*3 + ky
    int ix = idx & 63;
    int ic = r / 3, ky = r - ic * 3;
    int yy = y - 1 + ky;
    _Float16 v = (_Float16)0.f;
    if ((unsigned)yy < 64u) v = src[ic * 4096 + yy * 64 + ix];
    tile2[T2_DATA + r * 66 + ix] = v;
  }
#endif
  __syncthreads();
  // zero out-of-image rows (TDM path read garbage there)
  if (y == 0)
    for (int i = tid; i < 64 * 64; i += 256)
      tile2[T2_DATA + (i >> 6) * 3 * 66 + (i & 63)] = (_Float16)0.f;
  if (y == 63)
    for (int i = tid; i < 64 * 64; i += 256)
      tile2[T2_DATA + ((i >> 6) * 3 + 2) * 66 + (i & 63)] = (_Float16)0.f;
  __syncthreads();

  const int lane = tid & 31;
  const int wv   = tid >> 5;
  const int h    = lane >> 4;                         // half-wave
  const int rc   = wv & 3;                            // 16-pixel chunk of the row
  const int cgb  = (wv >> 2) * 4;                     // channel groups {0..3}/{4..7}
  const int m    = lane & 15;
  const int x    = rc * 16 + m;
  const v16h* wp = (const v16h*)wpack2;

  v8f acc0 = {}, acc1 = {}, acc2 = {}, acc3 = {};
  for (int ks = 0; ks < 32; ++ks) {                   // K' = 1024
    v16h a;
#pragma unroll
    for (int g = 0; g < 2; ++g) {
      int ic = ks * 2 + g;                            // lane-independent
      int baseA = T2_DATA + (ic * 3 + 2 * h) * 66 + x - 1;      // ky=2h: valid
      int baseB = (h == 0) ? (T2_DATA + (ic * 3 + 1) * 66 + x - 1)
                           : T2_ZSLOT;                // ky=2h+1: invalid if h==1
#pragma unroll
      for (int kx = 0; kx < 3; ++kx) {
        a[8 * g + kx]     = tile2[baseA + kx];
        a[8 * g + 4 + kx] = tile2[baseB + kx];
      }
      a[8 * g + 3] = (_Float16)0.f;                   // kx=3 always padding
      a[8 * g + 7] = (_Float16)0.f;
    }
    const int base = (ks * 8 + cgb) * 32 + lane;
    v16h b0 = wp[base +  0], b1 = wp[base + 32];
    v16h b2 = wp[base + 64], b3 = wp[base + 96];
    acc0 = WMMA_F16(a, b0, acc0);
    acc1 = WMMA_F16(a, b1, acc1);
    acc2 = WMMA_F16(a, b2, acc2);
    acc3 = WMMA_F16(a, b3, acc3);
  }

  const int n = lane & 15;
  const float gy = (float)y * (1.0f / 63.0f);         // linspace(0,1,64)
#pragma unroll
  for (int j = 0; j < 4; ++j) {
    v8f acc = (j == 0) ? acc0 : (j == 1) ? acc1 : (j == 2) ? acc2 : acc3;
    int ch = (cgb + j) * 16 + n;
    float bz = bias[ch];
    float w0 = pe_w[ch * 4 + 0], w1 = pe_w[ch * 4 + 1];
    float w2 = pe_w[ch * 4 + 2], w3 = pe_w[ch * 4 + 3];
    float embY = pe_b[ch] + gy * w0 + (1.0f - gy) * w2;
#pragma unroll
    for (int r8 = 0; r8 < 8; ++r8) {
      int px = rc * 16 + r8 + (h << 3);
      float gx  = (float)px * (1.0f / 63.0f);
      float emb = embY + gx * w1 + (1.0f - gx) * w3;
      float v   = fmaxf(acc[r8] + bz, 0.f) + emb;
      x2[((b * 64 + y) * 64 + px) * 128 + ch] = v;
    }
  }
}

// ---------------------------------------------------------------------------
// Kernel 3: masked spatial max. Block = (b,o); thread = channel (coalesced over
// channel-last x2). Mask is rois[b][o][::2][::2] (nearest, 1-byte jax bool);
// zeros from masked-out positions participate in the max, as in the reference.
// ---------------------------------------------------------------------------
__global__ __launch_bounds__(128)
void roi_max_kernel(const float* __restrict__ x2,
                    const unsigned char* __restrict__ rois,
                    float* __restrict__ out) {
  const int bo = blockIdx.x;                          // 0..255
  const int b = bo >> 5, o = bo & 31;
  const int c = threadIdx.x;                          // 0..127
  const unsigned char* mrow = rois + (size_t)(b * 32 + o) * 128 * 128;
  const float* xb = x2 + (size_t)b * 64 * 64 * 128;
  float acc = -INFINITY;
  for (int p = 0; p < 4096; ++p) {
    int yy = p >> 6, xx = p & 63;
    unsigned char mb = mrow[(2 * yy) * 128 + 2 * xx]; // wave-uniform broadcast
    float v = xb[(size_t)p * 128 + c];
    acc = fmaxf(acc, mb ? v : 0.0f);
  }
  out[(b * 32 + o) * 128 + c] = acc;
}

// ---------------------------------------------------------------------------
extern "C" void kernel_launch(void* const* d_in, const int* in_sizes, int n_in,
                              void* d_out, int out_size, void* d_ws, size_t ws_size,
                              hipStream_t stream) {
  const float*         images  = (const float*)d_in[0];
  const unsigned char* rois    = (const unsigned char*)d_in[1];
  /* d_in[2] = gt_pos, unused */
  const float*         conv1_w = (const float*)d_in[3];
  const float*         conv1_b = (const float*)d_in[4];
  const float*         conv2_w = (const float*)d_in[5];
  const float*         conv2_b = (const float*)d_in[6];
  const float*         pe_w    = (const float*)d_in[7];
  const float*         pe_b    = (const float*)d_in[8];

  // workspace layout (bytes):
  //   wpack1 @ 0        :  98304  (24*4*32*16 f16)
  //   wpack2 @ 98304    : 262144  (32*8*32*16 f16)
  //   pooled @ 360448   : 4 MiB   (8*64*64*64 f16)
  //   x2     @ 4554752  : 16 MiB  (8*64*64*128 f32)
  char* ws = (char*)d_ws;
  _Float16* wpack1 = (_Float16*)(ws);
  _Float16* wpack2 = (_Float16*)(ws + 98304);
  _Float16* pooled = (_Float16*)(ws + 360448);
  float*    x2     = (float*)   (ws + 4554752);

  pack_w1_kernel<<<192, 256, 0, stream>>>(conv1_w, wpack1);
  pack_w2_kernel<<<512, 256, 0, stream>>>(conv2_w, wpack2);
  conv1_pool_kernel<<<1024, 256, 0, stream>>>(images, wpack1, conv1_b, pooled);
  conv2_emb_kernel<<<512, 256, 0, stream>>>(pooled, wpack2, conv2_b, pe_w, pe_b, x2);
  roi_max_kernel<<<256, 128, 0, stream>>>(x2, rois, (float*)d_out);
}
